// WrappedCasualSelfAttention_88407606821429
// MI455X (gfx1250) — matrix operand order repaired and checked by reading, not compiled, not run on hardware
//
#include <hip/hip_runtime.h>

// ---------------------------------------------------------------------------
// Causal GQA self-attention block for MI455X (gfx1250, wave32, WMMA).
// Contractions via v_wmma_f32_16x16x32_f16; K/V double-buffered through LDS
// with async DMA (global_load_async_to_lds_b128 + s_wait_asynccnt pipeline).
// ---------------------------------------------------------------------------

typedef __attribute__((ext_vector_type(16))) _Float16 v16h;
typedef __attribute__((ext_vector_type(8)))  float    v8f;
typedef int v4i __attribute__((vector_size(4 * sizeof(int))));

#define E     64
#define QR    4
#define NKV   8
#define NH    (NKV * QR)      // 32
#define BATCH 2
#define SEQ   2048
#define ROWS  (BATCH * SEQ)   // 4096

#if defined(__has_builtin)
#if __has_builtin(__builtin_amdgcn_global_load_async_to_lds_b128) && \
    __has_builtin(__builtin_amdgcn_s_wait_asynccnt)
#define HAVE_ASYNC_LDS 1
#endif
#endif
#ifndef HAVE_ASYNC_LDS
#define HAVE_ASYNC_LDS 0
#endif

// Build explicit address-space pointers from generic ones via integers.
// Generic LDS pointer layout: [63:32]=aperture, [31:0]=LDS byte offset.
#define TO_GLOBAL_V4I(p) ((__attribute__((address_space(1))) v4i*)(unsigned long long)(p))
#define TO_LDS_V4I(p)    ((__attribute__((address_space(3))) v4i*)(unsigned int)(unsigned long long)(p))

__device__ __forceinline__ v8f wmma16(v16h a, v16h b, v8f c) {
  // (neg_a, A, neg_b, B, c_mod, C, reuse_a, reuse_b)
  return __builtin_amdgcn_wmma_f32_16x16x32_f16(false, a, false, b, (short)0, c,
                                                false, false);
}

// A fragment (16 rows x K=32, f16) from row-major f32, row stride ld.
// ISA 7.12.2: lane L -> row M=L&15; element e -> K = 16*(e>>3) + 8*(L>>4) + (e&7)
__device__ __forceinline__ v16h load_a_f32(const float* base, int ld, int lane,
                                           float scale) {
  const int m = lane & 15, h = lane >> 4;
  const float* r = base + (long)m * ld + h * 8;
  v16h a;
#pragma unroll
  for (int g = 0; g < 2; ++g)
#pragma unroll
    for (int t = 0; t < 8; ++t)
      a[g * 8 + t] = (_Float16)(r[g * 16 + t] * scale);
  return a;
}

// Same, from an LDS f16 tile.
__device__ __forceinline__ v16h load_a_lds(const _Float16* base, int ld, int lane) {
  const int m = lane & 15, h = lane >> 4;
  const _Float16* r = base + m * ld + h * 8;
  v16h a;
#pragma unroll
  for (int g = 0; g < 2; ++g)
#pragma unroll
    for (int t = 0; t < 8; ++t)
      a[g * 8 + t] = r[g * 16 + t];
  return a;
}

// B fragment (K=32 x N=16, f16) from row-major f32, row stride ld.
// Layout: lane L -> column N=L&15; element e -> K = 16*(e>>3) + 8*(L>>4) + (e&7).
__device__ __forceinline__ v16h load_b_f32(const float* base, int ld, int lane) {
  const float* r = base + (long)((lane >> 4) * 8) * ld + (lane & 15);
  v16h b;
#pragma unroll
  for (int e = 0; e < 16; ++e) b[e] = (_Float16)r[(long)(16 * (e >> 3) + (e & 7)) * ld];
  return b;
}

// ---------------------------------------------------------------------------
// Kernel 1: QKV projection. One wave -> 16x64 output tile (8 WMMAs),
// A fragments reused across 4 column tiles.
// grid.x = 256 row tiles, grid.y = 48 col groups (32 Q | 8 K | 8 V).
// ---------------------------------------------------------------------------
__global__ __launch_bounds__(32)
void qkv_kernel(const float* __restrict__ x,
                const float* __restrict__ wq, const float* __restrict__ bq,
                const float* __restrict__ wk, const float* __restrict__ bk,
                const float* __restrict__ wv, const float* __restrict__ bv,
                float* __restrict__ q_ws, float* __restrict__ k_ws,
                float* __restrict__ v_ws) {
  const int lane = threadIdx.x;
  const int row0 = blockIdx.x * 16;
  const int nt = blockIdx.y;

  const float* W; const float* bias; float* out; int ncols, n0;
  if (nt < 32)      { W = wq; bias = bq; out = q_ws; ncols = NH * E;  n0 = nt * 64; }
  else if (nt < 40) { W = wk; bias = bk; out = k_ws; ncols = NKV * E; n0 = (nt - 32) * 64; }
  else              { W = wv; bias = bv; out = v_ws; ncols = NKV * E; n0 = (nt - 40) * 64; }

  const int m = lane & 15, h = lane >> 4;

  const v16h a0 = load_a_f32(x + (long)row0 * E + 0,  E, lane, 1.0f);
  const v16h a1 = load_a_f32(x + (long)row0 * E + 32, E, lane, 1.0f);

  v8f acc[4];
#pragma unroll
  for (int t = 0; t < 4; ++t) {
    const float bv_ = bias[n0 + t * 16 + m];
#pragma unroll
    for (int r = 0; r < 8; ++r) acc[t][r] = bv_;
  }

#pragma unroll
  for (int t = 0; t < 4; ++t) {
    v16h b0 = load_b_f32(W + (long)0  * ncols + n0 + t * 16, ncols, lane);
    acc[t] = wmma16(a0, b0, acc[t]);
    v16h b1 = load_b_f32(W + (long)32 * ncols + n0 + t * 16, ncols, lane);
    acc[t] = wmma16(a1, b1, acc[t]);
  }

#pragma unroll
  for (int t = 0; t < 4; ++t)
#pragma unroll
    for (int r = 0; r < 8; ++r)
      out[(long)(row0 + r + 8 * h) * ncols + n0 + t * 16 + m] = acc[t][r];
}

// ---------------------------------------------------------------------------
// Kernel 2: RoPE, in place on Q and K workspaces.
// ---------------------------------------------------------------------------
__global__ void rope_kernel(float* __restrict__ q, float* __restrict__ k) {
  const long nq = (long)ROWS * NH * (E / 2);
  const long nk = (long)ROWS * NKV * (E / 2);
  const long total = nq + nk;
  for (long i = (long)blockIdx.x * blockDim.x + threadIdx.x; i < total;
       i += (long)gridDim.x * blockDim.x) {
    float* p; long idx; int nheads;
    if (i < nq) { p = q; idx = i;      nheads = NH;  }
    else        { p = k; idx = i - nq; nheads = NKV; }
    const int half = (int)(idx % (E / 2));
    const long srow = (idx / (E / 2)) / nheads;    // b*SEQ + s
    const int s = (int)(srow % SEQ);
    const float freq = __expf(-(float)(2 * half) * (9.210340371976184f / (float)E));
    const float ang = (float)s * freq;
    const float cs = __cosf(ang), sn = __sinf(ang);
    const float t1 = p[2 * idx], t2 = p[2 * idx + 1];
    p[2 * idx]     = t1 * cs - t2 * sn;
    p[2 * idx + 1] = t1 * sn + t2 * cs;
  }
}

// ---------------------------------------------------------------------------
// Kernel 3: causal flash attention.
// 4 waves / block, one (batch, head, 64-query strip) per block; each wave owns
// a 16-query tile. K/V tiles (32 keys) are DOUBLE-BUFFERED in LDS via async
// DMA: next tile's DMA overlaps current tile's WMMA work (in-order async
// completion => s_wait_asynccnt 8 releases exactly the current tile).
// ---------------------------------------------------------------------------
__global__ __launch_bounds__(128)
void attn_kernel(const float* __restrict__ q_ws, const float* __restrict__ k_ws,
                 const float* __restrict__ v_ws, float* __restrict__ o_ws) {
  __shared__ float    Ks[2][32 * 64];     // [buf][key][d] f32 (16 KB)
  __shared__ float    Vs[2][32 * 64];     // [buf][key][d] f32 (16 KB)
  __shared__ _Float16 pS[4][16 * 32];     // per-wave P transpose buffer (4 KB)

  const int tid  = threadIdx.x;
  const int lane = tid & 31;
  const int wave = tid >> 5;
  const int m = lane & 15, hh = lane >> 4;
  const int head = blockIdx.y, b = blockIdx.z;
  const int kvh = head >> 2;                       // GQA: 4 Q heads per KV head
  const int q0 = blockIdx.x * 64 + wave * 16;      // this wave's query tile

  const float* qbase = q_ws + ((long)(b * SEQ + q0) * NH + head) * E;
  // 1/sqrt(64) folded into Q fragment (exact in f16).
  const v16h aq0 = load_a_f32(qbase + 0,  NH * E, lane, 0.125f);
  const v16h aq1 = load_a_f32(qbase + 32, NH * E, lane, 0.125f);

  v8f oacc[4] = {};
  float mi[8], li[8];
#pragma unroll
  for (int r = 0; r < 8; ++r) { mi[r] = -1e30f; li[r] = 0.0f; }

  const int jmaxw = q0 + 15;                 // this wave's causal limit
  const int gjmax = blockIdx.x * 64 + 63;    // block-wide causal limit

  // Issue one KV tile (32 keys x 256 B for K and V) as 8 async b128 per wave.
  auto issue_tile = [&](int j, int p) {
    const float* kbase = k_ws + ((long)(b * SEQ + j) * NKV + kvh) * E;
    const float* vbase = v_ws + ((long)(b * SEQ + j) * NKV + kvh) * E;
#pragma unroll
    for (int it = 0; it < 4; ++it) {
      const int c = tid + it * 128;
      const int key = c >> 4, off = (c & 15) * 4;       // off in floats
      __builtin_amdgcn_global_load_async_to_lds_b128(
          TO_GLOBAL_V4I(kbase + (long)key * (NKV * E) + off),
          TO_LDS_V4I(&Ks[p][key * 64 + off]), 0, 0);
      __builtin_amdgcn_global_load_async_to_lds_b128(
          TO_GLOBAL_V4I(vbase + (long)key * (NKV * E) + off),
          TO_LDS_V4I(&Vs[p][key * 64 + off]), 0, 0);
    }
  };
  (void)issue_tile;

#if HAVE_ASYNC_LDS
  issue_tile(0, 0);                          // pipeline prologue
#endif

  int p = 0;
  for (int j = 0; j <= gjmax; j += 32, p ^= 1) {
#if HAVE_ASYNC_LDS
    const bool more = (j + 32) <= gjmax;     // block-uniform
    if (more) {
      issue_tile(j + 32, p ^ 1);             // overlap next DMA with compute
      __builtin_amdgcn_s_wait_asynccnt(8);   // oldest 8 (current tile) done
    } else {
      __builtin_amdgcn_s_wait_asynccnt(0);
    }
#else
    const float* kbase = k_ws + ((long)(b * SEQ + j) * NKV + kvh) * E;
    const float* vbase = v_ws + ((long)(b * SEQ + j) * NKV + kvh) * E;
#pragma unroll
    for (int it = 0; it < 4; ++it) {
      const int c = (tid + it * 128) * 4;               // element index
      const int key = c >> 6, off = c & 63;
      *(float4*)&Ks[p][key * 64 + off] = *(const float4*)&kbase[(long)key * (NKV * E) + off];
      *(float4*)&Vs[p][key * 64 + off] = *(const float4*)&vbase[(long)key * (NKV * E) + off];
    }
#endif
    __syncthreads();

    if (j <= jmaxw) {                        // wave-uniform: EXEC stays all-1s
      // scores for key tiles (j..j+15, j+16..j+31); contraction d=64 in 2 chunks
      v8f s0 = {}, s1 = {};
#pragma unroll
      for (int tile = 0; tile < 2; ++tile) {
        v8f* sp = tile ? &s1 : &s0;
#pragma unroll
        for (int c = 0; c < 2; ++c) {
          v16h bk;                           // lane = key column (m), elem = d (table order)
          const int d = c * 32 + 8 * hh;
#pragma unroll
          for (int e = 0; e < 16; ++e)
            bk[e] = (_Float16)Ks[p][(tile * 16 + m) * 64 + d + 16 * (e >> 3) + (e & 7)];
          *sp = wmma16(c ? aq1 : aq0, bk, *sp);
        }
      }

      // causal mask + online softmax (rows live on 16-lane halves)
      float al[8];
#pragma unroll
      for (int r = 0; r < 8; ++r) {
        const int qrow = q0 + r + 8 * hh;
        float v0 = (j + m      <= qrow) ? s0[r] : -1e30f;
        float v1 = (j + 16 + m <= qrow) ? s1[r] : -1e30f;
        float mx = fmaxf(v0, v1);
#pragma unroll
        for (int off = 8; off >= 1; off >>= 1)
          mx = fmaxf(mx, __shfl_xor(mx, off, 32));
        const float mnew = fmaxf(mi[r], mx);
        const float alpha = __expf(mi[r] - mnew);
        const float p0 = __expf(v0 - mnew);
        const float p1 = __expf(v1 - mnew);
        s0[r] = p0; s1[r] = p1;
        float ps = p0 + p1;
#pragma unroll
        for (int off = 8; off >= 1; off >>= 1)
          ps += __shfl_xor(ps, off, 32);
        li[r] = li[r] * alpha + ps;
        mi[r] = mnew;
        al[r] = alpha;
      }

#pragma unroll
      for (int t = 0; t < 4; ++t)
#pragma unroll
        for (int r = 0; r < 8; ++r) oacc[t][r] *= al[r];

      // P: C/D layout -> wave-private LDS -> A layout (in-order DS per wave)
      _Float16* pw = pS[wave];
#pragma unroll
      for (int r = 0; r < 8; ++r) {
        const int row = r + 8 * hh;
        pw[row * 32 + m]      = (_Float16)s0[r];
        pw[row * 32 + 16 + m] = (_Float16)s1[r];
      }
      asm volatile("s_wait_dscnt 0x0" ::: "memory");   // wave-local LDS ordering

      const v16h pa = load_a_lds(pw, 32, lane);
#pragma unroll
      for (int t = 0; t < 4; ++t) {
        v16h vb;                             // lane = d column (m), elem = key (table order)
        const float* r_ = &Vs[p][(8 * hh) * 64 + t * 16 + m];
#pragma unroll
        for (int e = 0; e < 16; ++e) vb[e] = (_Float16)r_[(16 * (e >> 3) + (e & 7)) * 64];
        oacc[t] = wmma16(pa, vb, oacc[t]);
      }
    }
    __syncthreads();                         // protect buffers before re-staging
  }

  float* obase = o_ws + ((long)(b * SEQ + q0) * NH + head) * E;
#pragma unroll
  for (int r = 0; r < 8; ++r) {
    const float inv = 1.0f / li[r];
#pragma unroll
    for (int t = 0; t < 4; ++t)
      obase[(long)(r + 8 * hh) * (NH * E) + t * 16 + m] = oacc[t][r] * inv;
  }
}

// ---------------------------------------------------------------------------
// Kernel 4: output projection (K=2048) + bias + residual + LayerNorm.
// ---------------------------------------------------------------------------
__global__ __launch_bounds__(32)
void oproj_ln_kernel(const float* __restrict__ o_ws, const float* __restrict__ wo,
                     const float* __restrict__ bo, const float* __restrict__ x,
                     const float* __restrict__ gamma, const float* __restrict__ beta,
                     float* __restrict__ out) {
  const int lane = threadIdx.x;
  const int m = lane & 15, hh = lane >> 4;
  const int row0 = blockIdx.x * 16;

  v8f acc[4];
#pragma unroll
  for (int t = 0; t < 4; ++t) {
    const int col = t * 16 + m;
    const float bv = bo[col];
#pragma unroll
    for (int r = 0; r < 8; ++r)
      acc[t][r] = bv + x[(long)(row0 + r + 8 * hh) * E + col];  // bias + residual
  }

  for (int k0 = 0; k0 < NH * E; k0 += 32) {
    if (k0 + 32 < NH * E)
      __builtin_prefetch(wo + (long)(k0 + 32) * E, 0, 1);
    v16h a = load_a_f32(o_ws + (long)row0 * (NH * E) + k0, NH * E, lane, 1.0f);
#pragma unroll
    for (int t = 0; t < 4; ++t) {
      v16h bfr = load_b_f32(wo + (long)k0 * E + t * 16, E, lane);
      acc[t] = wmma16(a, bfr, acc[t]);
    }
  }

  // LayerNorm over E=64: 4 in-register partials + 16-lane shuffle reduction.
#pragma unroll
  for (int r = 0; r < 8; ++r) {
    float s = acc[0][r] + acc[1][r] + acc[2][r] + acc[3][r];
    float q = acc[0][r] * acc[0][r] + acc[1][r] * acc[1][r] +
              acc[2][r] * acc[2][r] + acc[3][r] * acc[3][r];
#pragma unroll
    for (int off = 8; off >= 1; off >>= 1) {
      s += __shfl_xor(s, off, 32);
      q += __shfl_xor(q, off, 32);
    }
    const float mu  = s * (1.0f / E);
    const float var = q * (1.0f / E) - mu * mu;
    const float inv = rsqrtf(var + 1e-5f);
    const int row = row0 + r + 8 * hh;
#pragma unroll
    for (int t = 0; t < 4; ++t) {
      const int col = t * 16 + m;
      out[(long)row * E + col] = (acc[t][r] - mu) * inv * gamma[col] + beta[col];
    }
  }
}

// ---------------------------------------------------------------------------
extern "C" void kernel_launch(void* const* d_in, const int* in_sizes, int n_in,
                              void* d_out, int out_size, void* d_ws, size_t ws_size,
                              hipStream_t stream) {
  const float* x     = (const float*)d_in[0];
  const float* wq    = (const float*)d_in[1];
  const float* bq    = (const float*)d_in[2];
  const float* wk    = (const float*)d_in[3];
  const float* bk    = (const float*)d_in[4];
  const float* wv    = (const float*)d_in[5];
  const float* bv    = (const float*)d_in[6];
  const float* wo    = (const float*)d_in[7];
  const float* bo    = (const float*)d_in[8];
  const float* gamma = (const float*)d_in[9];
  const float* beta  = (const float*)d_in[10];
  float* out = (float*)d_out;

  // workspace layout (fp32): Q | K | V | attn-out   (~84 MB total)
  float* q_ws = (float*)d_ws;
  float* k_ws = q_ws + (size_t)ROWS * NH * E;
  float* v_ws = k_ws + (size_t)ROWS * NKV * E;
  float* o_ws = v_ws + (size_t)ROWS * NKV * E;

  qkv_kernel<<<dim3(ROWS / 16, 48), 32, 0, stream>>>(x, wq, bq, wk, bk, wv, bv,
                                                     q_ws, k_ws, v_ws);

  const long pairs = (long)ROWS * (NH + NKV) * (E / 2);
  rope_kernel<<<(int)((pairs + 255) / 256), 256, 0, stream>>>(q_ws, k_ws);

  attn_kernel<<<dim3(SEQ / 64, NH, BATCH), 128, 0, stream>>>(q_ws, k_ws, v_ws, o_ws);

  oproj_ln_kernel<<<ROWS / 16, 32, 0, stream>>>(o_ws, wo, bo, x, gamma, beta, out);
}
